// AxialAttention_54503134986450
// MI455X (gfx1250) — matrix operand order repaired and checked by reading, not compiled, not run on hardware
//
#include <hip/hip_runtime.h>
#include <hip/hip_bf16.h>

// ---------------------------------------------------------------------------
// AxialAttention for MI455X (gfx1250, wave32, WMMA f32_16x16x32_f16)
// B=2 S=2048 E=1024 H=16 D=64
// Each GEMM wave computes a 32x64 output tile (8 accumulators) so a 32-deep
// k-step needs 12 b128 loads for 8 WMMAs (vs 4 loads per WMMA unblocked).
// ---------------------------------------------------------------------------

#define Bsz 2
#define Ssz 2048
#define Esz 1024
#define Hsz 16
#define Dsz 64

typedef __attribute__((ext_vector_type(16))) _Float16 v16h;
typedef __attribute__((ext_vector_type(8)))  _Float16 v8h;
typedef __attribute__((ext_vector_type(8)))  float    v8f;

union V16H { v16h v; v8h h[2]; };

// A-fragment (16x32 f16, row-major source, leading dim = ld elements).
// ISA layout: lanes 0-15 hold M=lane, halfs = K{0..7, 16..23};
//             lanes 16-31 hold M=lane-16, halfs = K{8..15, 24..31}.
__device__ __forceinline__ v16h load_frag_a(const _Float16* __restrict__ tile,
                                            int ld, int lane) {
  int m  = lane & 15;
  int kb = (lane >> 4) << 3;            // 0 or 8
  const _Float16* p = tile + (size_t)m * ld + kb;
  V16H r;
  r.h[0] = *(const v8h*)(p);
  r.h[1] = *(const v8h*)(p + 16);
  return r.v;
}

// B-fragment for D = A x R^T: R is row-major (N x K), leading dim = ld.
// Lane = output column n (mod 16); lanes 0-15 hold K{0..7,16..23}, lanes 16-31 K{8..15,24..31}.
__device__ __forceinline__ v16h load_frag_b(const _Float16* __restrict__ tile,
                                            int ld, int lane) {
  int n  = lane & 15;
  int kb = (lane >> 4) << 3;            // 0 or 8
  const _Float16* p = tile + (size_t)n * ld + kb;
  V16H r;
  r.h[0] = *(const v8h*)(p);
  r.h[1] = *(const v8h*)(p + 16);
  return r.v;
}

__device__ __forceinline__ v8f wmma_f16(v16h a, v16h b, v8f c) {
  return __builtin_amdgcn_wmma_f32_16x16x32_f16(false, a, false, b,
                                                (short)0, c, false, false);
}

// One 32-deep k-step of a 32x64 register tile: 2 A frags, 4 B frags, 8 WMMAs.
__device__ __forceinline__ void mma_step(v8f (&acc)[8],
                                         const _Float16* __restrict__ A0,
                                         int lda,
                                         const _Float16* __restrict__ B0,
                                         int ldb, int lane) {
  v16h a0 = load_frag_a(A0, lda, lane);
  v16h a1 = load_frag_a(A0 + (size_t)16 * lda, lda, lane);
  v16h b0 = load_frag_b(B0, ldb, lane);
  v16h b1 = load_frag_b(B0 + (size_t)16 * ldb, ldb, lane);
  v16h b2 = load_frag_b(B0 + (size_t)32 * ldb, ldb, lane);
  v16h b3 = load_frag_b(B0 + (size_t)48 * ldb, ldb, lane);
  acc[0] = wmma_f16(a0, b0, acc[0]);
  acc[1] = wmma_f16(a0, b1, acc[1]);
  acc[2] = wmma_f16(a0, b2, acc[2]);
  acc[3] = wmma_f16(a0, b3, acc[3]);
  acc[4] = wmma_f16(a1, b0, acc[4]);
  acc[5] = wmma_f16(a1, b1, acc[5]);
  acc[6] = wmma_f16(a1, b2, acc[6]);
  acc[7] = wmma_f16(a1, b3, acc[7]);
}

// ---------------------------------------------------------------------------
__global__ void f32_to_f16_kernel(const float* __restrict__ in,
                                  _Float16* __restrict__ out, int n) {
  int i = blockIdx.x * blockDim.x + threadIdx.x;
  if (i < n) out[i] = (_Float16)in[i];
}

// ---------------------------------------------------------------------------
// QKV projection: OUT = Xh @ W^T + bias. blockIdx.z: 0=Q, 1=K, 2=V.
// Wave tile 32 rows x 64 cols; the 64-col block == exactly one head.
// Q,K stored [B,H,S,D]; V stored transposed [B,H,D,S] for the attn GEMM.
__global__ void qkv_proj_kernel(const _Float16* __restrict__ Xh,
                                const _Float16* __restrict__ Wqh,
                                const _Float16* __restrict__ Wkh,
                                const _Float16* __restrict__ Wvh,
                                const float* __restrict__ bq,
                                const float* __restrict__ bk,
                                const float* __restrict__ bv,
                                _Float16* __restrict__ Qh,
                                _Float16* __restrict__ Kh,
                                _Float16* __restrict__ Vt) {
  int lane = threadIdx.x & 31;
  int wave = threadIdx.x >> 5;
  int tile = blockIdx.x * 8 + wave;     // 2048 tiles: 128 row-tiles x 16 col-tiles
  int tr = tile >> 4;                   // [0,128) : 32-row tiles
  int tc = tile & 15;                   // [0,16)  : 64-col tiles == head index
  int z = blockIdx.z;
  const _Float16* W    = (z == 0) ? Wqh : (z == 1) ? Wkh : Wvh;
  const float*    bias = (z == 0) ? bq  : (z == 1) ? bk  : bv;

  const _Float16* Atile = Xh + (size_t)(tr * 32) * Esz;
  const _Float16* Btile = W  + (size_t)(tc * 64) * Esz;
  v8f acc[8] = {};
  for (int k0 = 0; k0 < Esz; k0 += 32)
    mma_step(acc, Atile + k0, Esz, Btile + k0, Esz, lane);

  int n  = lane & 15;
  int mb = (lane >> 4) * 8;
  int h  = tc;
#pragma unroll
  for (int mi = 0; mi < 2; ++mi) {
#pragma unroll
    for (int ni = 0; ni < 4; ++ni) {
      int d  = ni * 16 + n;
      float bval = bias[tc * 64 + d];
#pragma unroll
      for (int r = 0; r < 8; ++r) {
        int rr = tr * 32 + mi * 16 + mb + r;   // global row in [0, B*S)
        int b_ = rr >> 11;
        int s  = rr & 2047;
        _Float16 hv = (_Float16)(acc[mi * 4 + ni][r] + bval);
        if (z == 2) {
          Vt[(((size_t)b_ * Hsz + h) * Dsz + d) * Ssz + s] = hv;
        } else {
          _Float16* O = (z == 0) ? Qh : Kh;
          O[(((size_t)b_ * Hsz + h) * Ssz + s) * Dsz + d] = hv;
        }
      }
    }
  }
}

// ---------------------------------------------------------------------------
// scores[b,q,k] = mean_h( Q[b,h,q,:] . K[b,h,k,:] ) / sqrt(D)
// Wave tile 32(q) x 64(k); 16 heads x 2 k-steps = 32 mma_steps = 256 WMMAs.
__global__ void scores_kernel(const _Float16* __restrict__ Qh,
                              const _Float16* __restrict__ Kh,
                              float* __restrict__ scores) {
  int lane = threadIdx.x & 31;
  int wave = threadIdx.x >> 5;
  int tile = blockIdx.x * 8 + wave;     // 4096 = B * 64 * 32
  int b   = tile >> 11;
  int rem = tile & 2047;
  int qt  = rem >> 5;                   // [0,64)  : 32-row q tiles
  int kt  = rem & 31;                   // [0,32)  : 64-col k tiles

  v8f acc[8] = {};
  for (int h = 0; h < Hsz; ++h) {
    const _Float16* Qb = Qh + ((size_t)(b * Hsz + h) * Ssz + qt * 32) * Dsz;
    const _Float16* Kb = Kh + ((size_t)(b * Hsz + h) * Ssz + kt * 64) * Dsz;
    mma_step(acc, Qb, Dsz, Kb, Dsz, lane);
    mma_step(acc, Qb + 32, Dsz, Kb + 32, Dsz, lane);
  }

  const float sc = 1.0f / 128.0f;       // 1/(sqrt(64) * H)
  int n  = lane & 15;
  int mb = (lane >> 4) * 8;
  float* out = scores + (size_t)b * Ssz * Ssz +
               (size_t)(qt * 32) * Ssz + kt * 64;
#pragma unroll
  for (int mi = 0; mi < 2; ++mi)
#pragma unroll
    for (int ni = 0; ni < 4; ++ni)
#pragma unroll
      for (int r = 0; r < 8; ++r)
        out[(size_t)(mi * 16 + mb + r) * Ssz + ni * 16 + n] =
            acc[mi * 4 + ni][r] * sc;
}

// ---------------------------------------------------------------------------
// Row softmax over S=2048 columns; one 256-thread block per row; f16 output.
__global__ void softmax_kernel(const float* __restrict__ scores,
                               _Float16* __restrict__ wh) {
  __shared__ float red[256];
  int row = blockIdx.x;
  int tid = threadIdx.x;
  const float* s = scores + (size_t)row * Ssz;

  float v[8];
  float m = -3.402823e38f;
#pragma unroll
  for (int j = 0; j < 8; ++j) { v[j] = s[tid + j * 256]; m = fmaxf(m, v[j]); }
  red[tid] = m; __syncthreads();
  for (int st = 128; st > 0; st >>= 1) {
    if (tid < st) red[tid] = fmaxf(red[tid], red[tid + st]);
    __syncthreads();
  }
  m = red[0]; __syncthreads();

  float sum = 0.f;
#pragma unroll
  for (int j = 0; j < 8; ++j) { v[j] = __expf(v[j] - m); sum += v[j]; }
  red[tid] = sum; __syncthreads();
  for (int st = 128; st > 0; st >>= 1) {
    if (tid < st) red[tid] += red[tid + st];
    __syncthreads();
  }
  float inv = 1.0f / red[0];

  _Float16* o = wh + (size_t)row * Ssz;
#pragma unroll
  for (int j = 0; j < 8; ++j) o[tid + j * 256] = (_Float16)(v[j] * inv);
}

// ---------------------------------------------------------------------------
// attn[b,h,q,d] = sum_k weights[b,q,k] * V[b,h,k,d]   (B operand = Vt rows)
// Wave tile 32(q) x 64(d) == full head width; K loop over S = 64 mma_steps.
__global__ void attn_kernel(const _Float16* __restrict__ wh,
                            const _Float16* __restrict__ Vt,
                            _Float16* __restrict__ attn) {
  int lane = threadIdx.x & 31;
  int wave = threadIdx.x >> 5;
  int tile = blockIdx.x * 8 + wave;     // 2048 = B * H * 64
  int b   = tile >> 10;
  int rem = tile & 1023;
  int h   = rem >> 6;
  int qt  = rem & 63;                   // [0,64) : 32-row q tiles

  const _Float16* A = wh + (size_t)b * Ssz * Ssz + (size_t)(qt * 32) * Ssz;
  const _Float16* R = Vt + (size_t)(b * Hsz + h) * Dsz * Ssz;
  v8f acc[8] = {};
  for (int k0 = 0; k0 < Ssz; k0 += 32)
    mma_step(acc, A + k0, Ssz, R + k0, Ssz, lane);

  int n  = lane & 15;
  int mb = (lane >> 4) * 8;
  _Float16* O = attn + ((size_t)(b * Hsz + h) * Ssz + qt * 32) * Dsz;
#pragma unroll
  for (int mi = 0; mi < 2; ++mi)
#pragma unroll
    for (int ni = 0; ni < 4; ++ni)
#pragma unroll
      for (int r = 0; r < 8; ++r)
        O[(size_t)(mi * 16 + mb + r) * Dsz + ni * 16 + n] =
            (_Float16)acc[mi * 4 + ni][r];
}

// ---------------------------------------------------------------------------
// out = attn_flat @ Wo^T + bo  (f32 output). attn [B,H,S,D] contiguous == the
// reference's no-transpose reshape to [B*S, E].
__global__ void out_proj_kernel(const _Float16* __restrict__ attn,
                                const _Float16* __restrict__ Woh,
                                const float* __restrict__ bo,
                                float* __restrict__ out) {
  int lane = threadIdx.x & 31;
  int wave = threadIdx.x >> 5;
  int tile = blockIdx.x * 8 + wave;     // 2048 tiles: 128 x 16
  int tr = tile >> 4;
  int tc = tile & 15;

  const _Float16* Atile = attn + (size_t)(tr * 32) * Esz;
  const _Float16* Btile = Woh  + (size_t)(tc * 64) * Esz;
  v8f acc[8] = {};
  for (int k0 = 0; k0 < Esz; k0 += 32)
    mma_step(acc, Atile + k0, Esz, Btile + k0, Esz, lane);

  int n  = lane & 15;
  int mb = (lane >> 4) * 8;
#pragma unroll
  for (int mi = 0; mi < 2; ++mi) {
#pragma unroll
    for (int ni = 0; ni < 4; ++ni) {
      int cc = tc * 64 + ni * 16 + n;
      float bval = bo[cc];
#pragma unroll
      for (int r = 0; r < 8; ++r) {
        int rr = tr * 32 + mi * 16 + mb + r;
        out[(size_t)rr * Esz + cc] = acc[mi * 4 + ni][r] + bval;
      }
    }
  }
}

// ---------------------------------------------------------------------------
extern "C" void kernel_launch(void* const* d_in, const int* in_sizes, int n_in,
                              void* d_out, int out_size, void* d_ws, size_t ws_size,
                              hipStream_t stream) {
  const float* X  = (const float*)d_in[0];
  const float* Wq = (const float*)d_in[1];
  const float* bq = (const float*)d_in[2];
  const float* Wk = (const float*)d_in[3];
  const float* bk = (const float*)d_in[4];
  const float* Wv = (const float*)d_in[5];
  const float* bv = (const float*)d_in[6];
  const float* Wo = (const float*)d_in[7];
  const float* bo = (const float*)d_in[8];
  float* out = (float*)d_out;

  char* w = (char*)d_ws;
  size_t off = 0;
  auto alloc = [&](size_t bytes) -> void* {
    void* p = w + off;
    off += (bytes + 255) & ~(size_t)255;
    return p;
  };

  const size_t NX = (size_t)Bsz * Ssz * Esz;     // 4,194,304
  const size_t NW = (size_t)Esz * Esz;           // 1,048,576

  _Float16* Xh   = (_Float16*)alloc(NX * 2);
  _Float16* Wqh  = (_Float16*)alloc(NW * 2);
  _Float16* Wkh  = (_Float16*)alloc(NW * 2);
  _Float16* Wvh  = (_Float16*)alloc(NW * 2);
  _Float16* Woh  = (_Float16*)alloc(NW * 2);
  _Float16* Qh   = (_Float16*)alloc(NX * 2);
  _Float16* Kh   = (_Float16*)alloc(NX * 2);
  _Float16* Vt   = (_Float16*)alloc(NX * 2);
  _Float16* attn = (_Float16*)alloc(NX * 2);
  _Float16* wgt  = (_Float16*)alloc((size_t)Bsz * Ssz * Ssz * 2);
  float*    scr  = (float*)   alloc((size_t)Bsz * Ssz * Ssz * 4);
  (void)ws_size; (void)in_sizes; (void)n_in; (void)out_size;

  // fp32 -> fp16 conversions
  f32_to_f16_kernel<<<(int)((NX + 255) / 256), 256, 0, stream>>>(X, Xh, (int)NX);
  f32_to_f16_kernel<<<(int)((NW + 255) / 256), 256, 0, stream>>>(Wq, Wqh, (int)NW);
  f32_to_f16_kernel<<<(int)((NW + 255) / 256), 256, 0, stream>>>(Wk, Wkh, (int)NW);
  f32_to_f16_kernel<<<(int)((NW + 255) / 256), 256, 0, stream>>>(Wv, Wvh, (int)NW);
  f32_to_f16_kernel<<<(int)((NW + 255) / 256), 256, 0, stream>>>(Wo, Woh, (int)NW);

  // QKV projections (z selects Q/K/V)
  qkv_proj_kernel<<<dim3(256, 1, 3), 256, 0, stream>>>(
      Xh, Wqh, Wkh, Wvh, bq, bk, bv, Qh, Kh, Vt);

  // head-averaged scores
  scores_kernel<<<512, 256, 0, stream>>>(Qh, Kh, scr);

  // softmax -> f16 weights
  softmax_kernel<<<(int)(Bsz * Ssz), 256, 0, stream>>>(scr, wgt);

  // attention output
  attn_kernel<<<256, 256, 0, stream>>>(wgt, Vt, attn);

  // output projection
  out_proj_kernel<<<256, 256, 0, stream>>>(attn, Woh, bo, out);
}
